// SelfAttention_V1_6811818131541
// MI455X (gfx1250) — matrix operand order repaired and checked by reading, not compiled, not run on hardware
//
#include <hip/hip_runtime.h>

typedef __attribute__((ext_vector_type(16))) __bf16 v16bf;
typedef __attribute__((ext_vector_type(8)))  __bf16 v8bf;
typedef __attribute__((ext_vector_type(4)))  __bf16 v4bf;
typedef __attribute__((ext_vector_type(8)))  float  v8f;

#define N_TOK 4096
#define D_DIM 1024

static __device__ __forceinline__ v16bf cat8(v8bf lo, v8bf hi) {
  return __builtin_shufflevector(lo, hi, 0, 1, 2, 3, 4, 5, 6, 7,
                                         8, 9, 10, 11, 12, 13, 14, 15);
}
static __device__ __forceinline__ v8f wmma_bf16(v16bf a, v16bf b, v8f c) {
  return __builtin_amdgcn_wmma_f32_16x16x32_bf16(false, a, false, b,
                                                 (short)0, c, false, false);
}

// ---------------------------------------------------------------------------
// Kernel 1: C_bf16[4096x1024] = X f32 @ W f32  (blockIdx.z picks Wq/Wk/Wv)
// Block tile 128x128, 256 threads = 8 waves (4 along M x 2 along N),
// each wave: 2x4 grid of 16x16 WMMA tiles. K-step 64 through LDS so each
// stage feeds 16 WMMAs per wave between barriers.
// ---------------------------------------------------------------------------
#define TS 72  // padded LDS row stride (bf16 elems): 144B, 16B-aligned chunks,
               // dword-bank step 36 -> conflict-free for 16 lanes

__global__ __launch_bounds__(256)
void qkv_gemm_kernel(const float* __restrict__ X,
                     const float* __restrict__ Wq,
                     const float* __restrict__ Wk,
                     const float* __restrict__ Wv,
                     __bf16* __restrict__ Qb,
                     __bf16* __restrict__ Kb,
                     __bf16* __restrict__ Vb) {
  __shared__ __bf16 Xs[128 * TS];  // [m][k] bf16, 18 KB
  __shared__ __bf16 Wt[128 * TS];  // [n][k] bf16 (transposed W tile), 18 KB

  const int tid  = threadIdx.x;
  const int lane = tid & 31;
  const int ln   = lane & 15;
  const int g    = lane >> 4;
  const int wv   = tid >> 5;
  const int wm   = wv & 3;   // 4 waves along M
  const int wn   = wv >> 2;  // 2 waves along N
  const int m0   = blockIdx.y * 128;
  const int n0   = blockIdx.x * 128;
  const int z    = blockIdx.z;

  const float* W   = (z == 0) ? Wq : (z == 1) ? Wk : Wv;
  __bf16*      Out = (z == 0) ? Qb : (z == 1) ? Kb : Vb;

  v8f acc[2][4];
#pragma unroll
  for (int i = 0; i < 2; ++i)
#pragma unroll
    for (int j = 0; j < 4; ++j)
#pragma unroll
      for (int r = 0; r < 8; ++r) acc[i][j][r] = 0.0f;

  for (int k0 = 0; k0 < D_DIM; k0 += 64) {
    // Stage X tile (128x64 f32 -> bf16): 2048 float4 units, 8 per thread.
#pragma unroll
    for (int i = 0; i < 8; ++i) {
      int u   = tid + i * 256;
      int row = u >> 4;
      int c4  = (u & 15) * 4;
      const float4 v = *(const float4*)(X + (size_t)(m0 + row) * D_DIM + k0 + c4);
      v4bf p;
      p[0] = (__bf16)v.x; p[1] = (__bf16)v.y; p[2] = (__bf16)v.z; p[3] = (__bf16)v.w;
      *(v4bf*)(&Xs[row * TS + c4]) = p;
    }
    // Stage W tile (64x128) transposed into Wt[n][k]: 2048 units, 8/thread.
#pragma unroll
    for (int i = 0; i < 8; ++i) {
      int u   = tid + i * 256;
      int row = u >> 5;             // k index
      int c4  = (u & 31) * 4;       // n index base
      const float4 v = *(const float4*)(W + (size_t)(k0 + row) * D_DIM + n0 + c4);
      Wt[(c4 + 0) * TS + row] = (__bf16)v.x;
      Wt[(c4 + 1) * TS + row] = (__bf16)v.y;
      Wt[(c4 + 2) * TS + row] = (__bf16)v.z;
      Wt[(c4 + 3) * TS + row] = (__bf16)v.w;
    }
    __syncthreads();

#pragma unroll
    for (int d0 = 0; d0 < 64; d0 += 32) {
      // A fragments: lane = row m; elems K = d0 + 16g + e contiguous
      v16bf a[2];
#pragma unroll
      for (int im = 0; im < 2; ++im) {
        int m   = wm * 32 + im * 16 + ln;
        v8bf lo = *(const v8bf*)(&Xs[m * TS + d0 + 16 * g]);
        v8bf hi = *(const v8bf*)(&Xs[m * TS + d0 + 16 * g + 8]);
        a[im]   = cat8(lo, hi);
      }
      // B fragments: lane = column n; elems K = d0 + 16g + e contiguous
#pragma unroll
      for (int in = 0; in < 4; ++in) {
        int n    = wn * 64 + in * 16 + ln;
        v8bf blo = *(const v8bf*)(&Wt[n * TS + d0 + 16 * g]);
        v8bf bhi = *(const v8bf*)(&Wt[n * TS + d0 + 16 * g + 8]);
        v16bf b  = cat8(blo, bhi);
        acc[0][in] = wmma_bf16(a[0], b, acc[0][in]);
        acc[1][in] = wmma_bf16(a[1], b, acc[1][in]);
      }
    }
    __syncthreads();
  }

  // C/D layout: element (reg r, lane) -> row = r + 8g, col = ln
#pragma unroll
  for (int im = 0; im < 2; ++im)
#pragma unroll
    for (int in = 0; in < 4; ++in)
#pragma unroll
      for (int r = 0; r < 8; ++r) {
        int m = m0 + wm * 32 + im * 16 + r + 8 * g;
        int n = n0 + wn * 64 + in * 16 + ln;
        Out[(size_t)m * D_DIM + n] = (__bf16)acc[im][in][r];
      }
}

// ---------------------------------------------------------------------------
// Kernel 2: Vt[1024][4096] = transpose(V[4096][1024]), bf16, LDS tiled.
// ---------------------------------------------------------------------------
__global__ __launch_bounds__(256)
void transpose_v_kernel(const __bf16* __restrict__ src,
                        __bf16* __restrict__ dst) {
  __shared__ __bf16 tile[32][33];
  const int tx = threadIdx.x;  // 0..31
  const int ty = threadIdx.y;  // 0..7
  const int c0 = blockIdx.x * 32;  // column (d) base in src
  const int r0 = blockIdx.y * 32;  // row (token) base in src
  for (int i = ty; i < 32; i += 8)
    tile[i][tx] = src[(size_t)(r0 + i) * D_DIM + c0 + tx];
  __syncthreads();
  for (int i = ty; i < 32; i += 8)
    dst[(size_t)(c0 + i) * N_TOK + r0 + tx] = tile[tx][i];
}

// ---------------------------------------------------------------------------
// Kernel 3: flash attention. One workgroup = 16 query rows, 8 waves.
// Key blocks of 128: wave w computes S tile cols [16w,16w+16); for P@V wave w
// owns head-dim slice [128w,128w+128) (8 x v8f accumulator). Online softmax
// merged across waves via LDS (2 barriers / key block). Next key block's K and
// V^T rows are prefetched (global_prefetch_b8) while this block computes.
// ---------------------------------------------------------------------------
#define QS_STRIDE 1032  // 1024 + 8 pad (keeps 16B alignment, breaks banks)
#define PS_STRIDE 136   // 128 + 8 pad

__global__ __launch_bounds__(256)
void flash_attn_kernel(const __bf16* __restrict__ Qb,
                       const __bf16* __restrict__ Kb,
                       const __bf16* __restrict__ Vt,
                       float* __restrict__ out) {
  __shared__ __bf16 Qs[16 * QS_STRIDE];  // 33 KB
  __shared__ __bf16 Ps[16 * PS_STRIDE];  // 4.25 KB
  __shared__ float  wmax[8 * 16];
  __shared__ float  wsum[8 * 16];
  __shared__ float  m_run[16];
  __shared__ float  l_run[16];

  const int tid  = threadIdx.x;
  const int lane = tid & 31;
  const int ln   = lane & 15;
  const int g    = lane >> 4;
  const int wv   = tid >> 5;
  const int q0   = blockIdx.x * 16;

  // Stage the 16x1024 Q block into LDS (bf16, 16B vector copies).
#pragma unroll
  for (int i = 0; i < 8; ++i) {
    int u  = tid + i * 256;  // 0..2047 chunks of 8 elems
    int m  = u >> 7;
    int c8 = (u & 127) * 8;
    *(v8bf*)(&Qs[m * QS_STRIDE + c8]) =
        *(const v8bf*)(Qb + (size_t)(q0 + m) * D_DIM + c8);
  }
  if (tid < 16) { m_run[tid] = -1e30f; l_run[tid] = 0.0f; }
  __syncthreads();

  v8f acc[8];
#pragma unroll
  for (int t = 0; t < 8; ++t)
#pragma unroll
    for (int r = 0; r < 8; ++r) acc[t][r] = 0.0f;

  const float scale = 0.03125f;  // 1/sqrt(1024)

  for (int j = 0; j < 32; ++j) {
    const int key = j * 128 + wv * 16 + ln;  // this lane's key column

    // ---- S = Q @ K^T (16x16 tile per wave), K-dim 1024 in 32 WMMA steps
    v8f s;
#pragma unroll
    for (int r = 0; r < 8; ++r) s[r] = 0.0f;
#pragma unroll 4
    for (int d0 = 0; d0 < D_DIM; d0 += 32) {
      // Prefetch next key block's row for this lane (one per 128B line).
      if (j < 31 && (d0 & 63) == 0)
        __builtin_prefetch(Kb + (size_t)(key + 128) * D_DIM + d0, 0, 3);
      v8bf lo = *(const v8bf*)(&Qs[ln * QS_STRIDE + d0 + 16 * g]);
      v8bf hi = *(const v8bf*)(&Qs[ln * QS_STRIDE + d0 + 16 * g + 8]);
      v16bf a = cat8(lo, hi);
      v16bf b = *(const v16bf*)(Kb + (size_t)key * D_DIM + d0 + 16 * g);
      s = wmma_bf16(a, b, s);
    }
    float sv[8];
#pragma unroll
    for (int r = 0; r < 8; ++r) sv[r] = s[r] * scale;

    // ---- per-wave row max (reduce over the 16 lanes of each half)
#pragma unroll
    for (int r = 0; r < 8; ++r) {
      float v = sv[r];
      v = fmaxf(v, __shfl_xor(v, 1));
      v = fmaxf(v, __shfl_xor(v, 2));
      v = fmaxf(v, __shfl_xor(v, 4));
      v = fmaxf(v, __shfl_xor(v, 8));
      if (ln == 0) wmax[wv * 16 + 8 * g + r] = v;
    }
    __syncthreads();  // barrier 1: wmax complete

    // ---- new running max + rescale factor (computed redundantly per wave)
    float mnew[8], alpha[8];
#pragma unroll
    for (int r = 0; r < 8; ++r) {
      int row   = 8 * g + r;
      float old = m_run[row];
      float t   = old;
#pragma unroll
      for (int w2 = 0; w2 < 8; ++w2) t = fmaxf(t, wmax[w2 * 16 + row]);
      mnew[r]  = t;
      alpha[r] = __expf(old - t);
    }

    // ---- P = exp(S - mnew): row sums, bf16 P tile to LDS, rescale O
#pragma unroll
    for (int r = 0; r < 8; ++r) {
      float p = __expf(sv[r] - mnew[r]);
      Ps[(8 * g + r) * PS_STRIDE + wv * 16 + ln] = (__bf16)p;
      float v = p;
      v += __shfl_xor(v, 1);
      v += __shfl_xor(v, 2);
      v += __shfl_xor(v, 4);
      v += __shfl_xor(v, 8);
      if (ln == 0) wsum[wv * 16 + 8 * g + r] = v;
#pragma unroll
      for (int t = 0; t < 8; ++t) acc[t][r] *= alpha[r];
    }
    __syncthreads();  // barrier 2: Ps/wsum complete, prev Ps readers done

    // ---- wave 0 advances running stats (uses its own mnew/alpha registers)
    if (wv == 0 && ln == 0) {
#pragma unroll
      for (int r = 0; r < 8; ++r) {
        int row   = 8 * g + r;
        float sum = 0.0f;
#pragma unroll
        for (int w2 = 0; w2 < 8; ++w2) sum += wsum[w2 * 16 + row];
        l_run[row] = l_run[row] * alpha[r] + sum;
        m_run[row] = mnew[r];
      }
    }

    // ---- O += P @ V : contraction over 128 keys (4 WMMA k-steps), wave's
    //      head-dim slice is [128*wv, 128*wv+128) = 8 tiles of 16 columns.
#pragma unroll
    for (int kk = 0; kk < 4; ++kk) {
      v8bf plo = *(const v8bf*)(&Ps[ln * PS_STRIDE + kk * 32 + 16 * g]);
      v8bf phi = *(const v8bf*)(&Ps[ln * PS_STRIDE + kk * 32 + 16 * g + 8]);
      v16bf pa = cat8(plo, phi);
#pragma unroll
      for (int t = 0; t < 8; ++t) {
        const int dcol = wv * 128 + t * 16 + ln;
        // Prefetch next key block's V^T span for this row (2 x 128B halves).
        if (kk == 0 && j < 31)
          __builtin_prefetch(Vt + (size_t)dcol * N_TOK + (j + 1) * 128 + 64 * g,
                             0, 3);
        v16bf b = *(const v16bf*)(Vt + (size_t)dcol * N_TOK + j * 128 +
                                  kk * 32 + 16 * g);
        acc[t] = wmma_bf16(pa, b, acc[t]);
      }
    }
  }
  __syncthreads();  // l_run final values visible

  float linv[8];
#pragma unroll
  for (int r = 0; r < 8; ++r) linv[r] = 1.0f / l_run[8 * g + r];
#pragma unroll
  for (int t = 0; t < 8; ++t)
#pragma unroll
    for (int r = 0; r < 8; ++r)
      out[(size_t)(q0 + 8 * g + r) * D_DIM + wv * 128 + t * 16 + ln] =
          acc[t][r] * linv[r];
}

// ---------------------------------------------------------------------------
extern "C" void kernel_launch(void* const* d_in, const int* in_sizes, int n_in,
                              void* d_out, int out_size, void* d_ws,
                              size_t ws_size, hipStream_t stream) {
  (void)in_sizes; (void)n_in; (void)out_size; (void)ws_size;
  const float* x  = (const float*)d_in[0];
  const float* wq = (const float*)d_in[1];
  const float* wk = (const float*)d_in[2];
  const float* wv = (const float*)d_in[3];
  float* out = (float*)d_out;

  char* ws = (char*)d_ws;
  const size_t seg = (size_t)N_TOK * D_DIM * sizeof(__bf16);  // 8 MB
  __bf16* Qb = (__bf16*)(ws);
  __bf16* Kb = (__bf16*)(ws + seg);
  __bf16* Vb = (__bf16*)(ws + 2 * seg);
  __bf16* Vt = (__bf16*)(ws + 3 * seg);

  qkv_gemm_kernel<<<dim3(D_DIM / 128, N_TOK / 128, 3), 256, 0, stream>>>(
      x, wq, wk, wv, Qb, Kb, Vb);
  transpose_v_kernel<<<dim3(D_DIM / 32, N_TOK / 32), dim3(32, 8), 0, stream>>>(
      Vb, Vt);
  flash_attn_kernel<<<N_TOK / 16, 256, 0, stream>>>(Qb, Kb, Vt, out);
}
